// Attention_16149077033012
// MI455X (gfx1250) — matrix operand order repaired and checked by reading, not compiled, not run on hardware
//
#include <hip/hip_runtime.h>
#include <hip/hip_bf16.h>

#define B_  64
#define N_  4096
#define D_  512
#define H_  512
#define NCH 8          // N-split chunks for the features reduction

typedef __attribute__((ext_vector_type(16))) __bf16 v16bf;
typedef __attribute__((ext_vector_type(8)))  __bf16 v8bf;
typedef __attribute__((ext_vector_type(8)))  float  v8f;

__device__ __forceinline__ unsigned short f2bf_rne(float x) {
    unsigned int u = __float_as_uint(x);
    u += 0x7FFFu + ((u >> 16) & 1u);
    return (unsigned short)(u >> 16);
}

// Hardware V_TANH_F32 (gfx1250 TRANS op): co-executes with WMMA.
__device__ __forceinline__ float fast_tanh(float x) {
#if __has_builtin(__builtin_amdgcn_tanhf)
    return __builtin_amdgcn_tanhf(x);
#else
    return tanhf(x);
#endif
}

// ---------------------------------------------------------------------------
// Kernel 1: convert Wc (H x D fp32) -> bf16 into workspace (stays L2-resident)
// ---------------------------------------------------------------------------
__global__ __launch_bounds__(256)
void wc_to_bf16(const float* __restrict__ wc, unsigned int* __restrict__ out) {
    int i = blockIdx.x * 256 + threadIdx.x;     // over H*D/2 packed pairs
    float2 f = ((const float2*)wc)[i];
    out[i] = (unsigned int)f2bf_rne(f.x) | ((unsigned int)f2bf_rne(f.y) << 16);
}

// ---------------------------------------------------------------------------
// Kernel 2: keys[b,h] = dot(h[b,:], Wh[h,:])   (tiny: 33 MFLOP total)
// ---------------------------------------------------------------------------
__global__ __launch_bounds__(256)
void keys_kernel(const float* __restrict__ hin, const float* __restrict__ Wh,
                 float* __restrict__ keys) {
    int idx = blockIdx.x * 256 + threadIdx.x;   // B*H threads
    int b  = idx >> 9;
    int hh = idx & (H_ - 1);
    const float4* hv = (const float4*)(hin + (size_t)b  * D_);
    const float4* wv = (const float4*)(Wh  + (size_t)hh * D_);
    float acc = 0.f;
#pragma unroll 4
    for (int i = 0; i < D_ / 4; ++i) {
        float4 a = hv[i], w = wv[i];
        acc += a.x * w.x + a.y * w.y + a.z * w.z + a.w * w.w;
    }
    keys[idx] = acc;
}

// ---------------------------------------------------------------------------
// Kernel 3: aff[b,n] = sum_h tanh(keys[b,h] + cand[b,n,:]·Wc[h,:]) * Wout[h]
// bf16 WMMA GEMM. A tile (64x512 bf16) staged once in 64KB LDS (XOR swizzle),
// then hoisted by the compiler into 128 resident VGPRs; inner loop is pure
// global B-fragment loads (L2) + v_wmma_f32_16x16x32_bf16.
// ---------------------------------------------------------------------------
__global__ __launch_bounds__(128)
void aff_kernel(const float* __restrict__ cand, const unsigned short* __restrict__ wcb,
                const float* __restrict__ keys, const float* __restrict__ wout,
                float* __restrict__ aff) {
    __shared__ uint4 ldsA[4096];                // 64 rows x 64 chunks of 16B = 64KB

    const int b     = blockIdx.y;
    const int nbase = blockIdx.x * 64;
    const int tid   = threadIdx.x;

    // ---- stage candidates tile 64x512 fp32 -> bf16 LDS (swizzled 16B chunks)
    const float4* src = (const float4*)(cand + ((size_t)b * N_ + nbase) * D_);
    uint2* lds2 = (uint2*)ldsA;
    for (int i = 0; i < 64; ++i) {
        int cid = i * 128 + tid;                // 8192 float4 quads total
        int row = cid >> 7;                     // 128 quads per row
        int c4  = cid & 127;
        float4 f = src[(size_t)row * 128 + c4];
        unsigned int lo = (unsigned int)f2bf_rne(f.x) | ((unsigned int)f2bf_rne(f.y) << 16);
        unsigned int hi = (unsigned int)f2bf_rne(f.z) | ((unsigned int)f2bf_rne(f.w) << 16);
        int chunk = c4 >> 1;                    // 8-element (16B) chunk index
        int sub   = c4 & 1;
        lds2[(row * 64 + (chunk ^ (row & 7))) * 2 + sub] = make_uint2(lo, hi);
    }
    __syncthreads();

    const int lane = tid & 31;
    const int mtb  = (tid >> 5) * 16;           // wave's 16-row M tile
    const int nloc = lane & 15;                 // column (h) within tile / A row
    const int grp  = lane >> 4;                 // K-half select (contiguous k order used here)
    const int arow = mtb + nloc;                // A-matrix row held by this lane
    const int aswz = arow & 7;

    // A-fragment LDS byte addresses are invariant across the whole h-loop:
    // precompute all 16x2 of them once (kept in VGPRs by the full unroll).
    unsigned aAddr0[16], aAddr1[16];
#pragma unroll
    for (int kt = 0; kt < D_ / 32; ++kt) {
        const int kchunk = kt * 4;
        aAddr0[kt] = (unsigned)((arow * 64 + ((kchunk + 2 * grp)     ^ aswz)) * 16);
        aAddr1[kt] = (unsigned)((arow * 64 + ((kchunk + 2 * grp + 1) ^ aswz)) * 16);
    }
    const char* ldsb = (const char*)ldsA;

    float afin[8];
#pragma unroll
    for (int r = 0; r < 8; ++r) afin[r] = 0.f;

    for (int ht = 0; ht < H_ / 16; ++ht) {
        const int hcol = ht * 16 + nloc;
        const float kv = keys[b * H_ + hcol];
        const float wv = wout[hcol];
        const unsigned short* bp = wcb + (size_t)hcol * D_ + grp * 16;

        v8f c = {};
#pragma unroll
        for (int kt = 0; kt < D_ / 32; ++kt) {
            // A fragment (16x32 bf16): lanes 0-15 hold K 0-15,
            // lanes 16-31 hold K 16-31 of their row.
            v8bf alo = *(const v8bf*)(ldsb + aAddr0[kt]);
            v8bf ahi = *(const v8bf*)(ldsb + aAddr1[kt]);
            v16bf a = __builtin_shufflevector(alo, ahi,
                        0,1,2,3,4,5,6,7,8,9,10,11,12,13,14,15);
            // B fragment (32x16 bf16): lane holds one h-column, 16 contiguous K.
            v16bf bf = *(const v16bf*)(bp + kt * 32);
            c = __builtin_amdgcn_wmma_f32_16x16x32_bf16(
                    false, a, false, bf, (short)0, c, false, false);
        }
        // C layout: VGPR r -> M = r (+8 for lanes 16-31), lane%16 -> N (= h col)
#pragma unroll
        for (int r = 0; r < 8; ++r)
            afin[r] += fast_tanh(kv + c[r]) * wv;
    }

    // reduce over the 16 h-columns held across the 16-lane group
#pragma unroll
    for (int r = 0; r < 8; ++r) {
        float v = afin[r];
        v += __shfl_xor(v, 1, 16);
        v += __shfl_xor(v, 2, 16);
        v += __shfl_xor(v, 4, 16);
        v += __shfl_xor(v, 8, 16);
        afin[r] = v;
    }
    if (nloc == 0) {
        const int rowbase = nbase + mtb + grp * 8;
#pragma unroll
        for (int r = 0; r < 8; ++r)
            aff[(size_t)b * N_ + rowbase + r] = afin[r];
    }
}

// ---------------------------------------------------------------------------
// Kernel 4: softmax over N per batch; writes weights into its d_out slice
// ---------------------------------------------------------------------------
__global__ __launch_bounds__(256)
void softmax_kernel(const float* __restrict__ aff, const unsigned char* __restrict__ mask,
                    float* __restrict__ weights) {
    const int b = blockIdx.x, tid = threadIdx.x;
    __shared__ float redmax[8];
    __shared__ float redsum[8];

    float v[16];
    float mx = -3.4e38f;
#pragma unroll
    for (int i = 0; i < 16; ++i) {
        int n = tid + i * 256;
        float a = aff[(size_t)b * N_ + n];
        if (mask[(size_t)b * N_ + n]) a = -3.4e38f;
        v[i] = a;
        mx = fmaxf(mx, a);
    }
#pragma unroll
    for (int off = 16; off; off >>= 1) mx = fmaxf(mx, __shfl_xor(mx, off, 32));
    if ((tid & 31) == 0) redmax[tid >> 5] = mx;
    __syncthreads();
    mx = redmax[0];
#pragma unroll
    for (int w = 1; w < 8; ++w) mx = fmaxf(mx, redmax[w]);

    float s = 0.f;
#pragma unroll
    for (int i = 0; i < 16; ++i) { float e = __expf(v[i] - mx); v[i] = e; s += e; }
#pragma unroll
    for (int off = 16; off; off >>= 1) s += __shfl_xor(s, off, 32);
    if ((tid & 31) == 0) redsum[tid >> 5] = s;
    __syncthreads();
    s = 0.f;
#pragma unroll
    for (int w = 0; w < 8; ++w) s += redsum[w];
    const float inv = 1.f / s;
#pragma unroll
    for (int i = 0; i < 16; ++i) {
        int n = tid + i * 256;
        weights[(size_t)b * N_ + n] = v[i] * inv;
    }
}

// ---------------------------------------------------------------------------
// Kernel 5a: partial features over an N-chunk (split 8x for HBM concurrency).
// Deep unroll + global_prefetch keeps ~16 independent loads in flight/thread.
// ---------------------------------------------------------------------------
__global__ __launch_bounds__(256)
void features_partial(const float* __restrict__ cand, const float* __restrict__ weights,
                      float* __restrict__ part) {
    const int b  = blockIdx.z;
    const int nc = blockIdx.y;
    const int d  = blockIdx.x * 256 + threadIdx.x;
    const int n0 = nc * (N_ / NCH);
    const float* cp = cand + ((size_t)b * N_ + n0) * D_ + d;
    const float* wp = weights + (size_t)b * N_ + n0;
    float acc = 0.f;
#pragma unroll 16
    for (int n = 0; n < N_ / NCH; ++n) {
        if ((n & 15) == 0 && n + 32 < N_ / NCH)
            __builtin_prefetch(cp + (size_t)(n + 32) * D_, 0, 0);  // global_prefetch_b8
        acc += wp[n] * cp[(size_t)n * D_];
    }
    part[((size_t)b * NCH + nc) * D_ + d] = acc;
}

// ---------------------------------------------------------------------------
// Kernel 5b: reduce the NCH partials -> features
// ---------------------------------------------------------------------------
__global__ __launch_bounds__(256)
void features_reduce(const float* __restrict__ part, float* __restrict__ features) {
    const int idx = blockIdx.x * 256 + threadIdx.x;  // B*D threads
    const int b = idx >> 9;
    const int d = idx & (D_ - 1);
    float acc = 0.f;
#pragma unroll
    for (int nc = 0; nc < NCH; ++nc)
        acc += part[((size_t)b * NCH + nc) * D_ + d];
    features[idx] = acc;
}

// ---------------------------------------------------------------------------
extern "C" void kernel_launch(void* const* d_in, const int* in_sizes, int n_in,
                              void* d_out, int out_size, void* d_ws, size_t ws_size,
                              hipStream_t stream) {
    const float*         hin  = (const float*)d_in[0];          // (B, D)
    const float*         cand = (const float*)d_in[1];          // (B, N, D)
    const unsigned char* mask = (const unsigned char*)d_in[2];  // (B, N) bool
    const float*         Wh   = (const float*)d_in[3];          // (H, D)
    const float*         Wc   = (const float*)d_in[4];          // (H, D)
    const float*         Wout = (const float*)d_in[5];          // (H,)

    char* ws = (char*)d_ws;
    size_t off = 0;
    float*          keys = (float*)(ws + off);          off += (size_t)B_ * H_ * 4;       // 128KB
    unsigned short* wcb  = (unsigned short*)(ws + off); off += (size_t)H_ * D_ * 2;       // 512KB
    float*          aff  = (float*)(ws + off);          off += (size_t)B_ * N_ * 4;       // 1MB
    float*          part = (float*)(ws + off);          off += (size_t)B_ * NCH * D_ * 4; // 1MB

    float* features = (float*)d_out;                    // first B*D outputs
    float* weights  = (float*)d_out + (size_t)B_ * D_;  // then B*N outputs

    hipLaunchKernelGGL(wc_to_bf16,      dim3((H_ * D_ / 2) / 256), dim3(256), 0, stream,
                       Wc, (unsigned int*)wcb);
    hipLaunchKernelGGL(keys_kernel,     dim3((B_ * H_) / 256),     dim3(256), 0, stream,
                       hin, Wh, keys);
    hipLaunchKernelGGL(aff_kernel,      dim3(N_ / 64, B_),         dim3(128), 0, stream,
                       cand, wcb, keys, Wout, aff);
    hipLaunchKernelGGL(softmax_kernel,  dim3(B_),                  dim3(256), 0, stream,
                       aff, mask, weights);
    hipLaunchKernelGGL(features_partial,dim3(D_ / 256, NCH, B_),   dim3(256), 0, stream,
                       cand, weights, part);
    hipLaunchKernelGGL(features_reduce, dim3((B_ * D_) / 256),     dim3(256), 0, stream,
                       part, features);
}
